// DeepseekV2Attention_52785148067909
// MI455X (gfx1250) — matrix operand order repaired and checked by reading, not compiled, not run on hardware
//
#include <hip/hip_runtime.h>

// ---------------------------------------------------------------------------
// DeepSeek-V2 MLA attention forward for gfx1250 (MI455X), wave32 + WMMA +
// async global->LDS staging (ASYNCcnt) with double-buffered GEMM pipeline.
// ---------------------------------------------------------------------------

constexpr int kT      = 2048;
constexpr int kH      = 32;
constexpr int kHidden = 5120;
constexpr int kQLora  = 1536;
constexpr int kKVLora = 512;
constexpr int kDNope  = 128;
constexpr int kDRope  = 64;
constexpr int kDQK    = 192;   // 128 + 64
constexpr int kDV     = 128;
constexpr float kScaling = 0.07216878364870323f;   // 192^-0.5
constexpr float kEps  = 1e-6f;

typedef __attribute__((ext_vector_type(16))) _Float16 v16h;
typedef __attribute__((ext_vector_type(8)))  _Float16 v8h;
typedef __attribute__((ext_vector_type(8)))  float    v8f;

__device__ __forceinline__ v8f vzero8() {
  v8f z;
#pragma unroll
  for (int i = 0; i < 8; ++i) z[i] = 0.0f;
  return z;
}

__device__ __forceinline__ v8f wmma_f16(v16h a, v16h b, v8f c) {
  // (neg_a, A, neg_b, B, c_mod, C, reuse_a, reuse_b)
  return __builtin_amdgcn_wmma_f32_16x16x32_f16(false, a, false, b, (short)0, c,
                                                false, false);
}

// ---------------------------------------------------------------------------
// Async global->LDS copy (16 bytes per lane), GV addressing mode.
// The low 32 bits of a generic LDS pointer are the LDS aperture offset
// (ISA 10.2), which is what VDST must hold.
// Tracked by ASYNCcnt; pair with wait_async0() before the consuming barrier.
// ---------------------------------------------------------------------------
__device__ __forceinline__ void async_copy_b128(void* lds_dst, const void* gsrc) {
  asm volatile("global_load_async_to_lds_b128 %0, %1, off"
               :
               : "v"((unsigned int)(unsigned long long)(uintptr_t)lds_dst),
                 "v"((unsigned long long)(uintptr_t)gsrc)
               : "memory");
}

__device__ __forceinline__ void wait_async0() {
#if __has_builtin(__builtin_amdgcn_s_wait_asynccnt)
  __builtin_amdgcn_s_wait_asynccnt(0);
#else
  asm volatile("s_wait_asynccnt 0" ::: "memory");
#endif
}

// A fragment (16x32 f16, M x K). Per ISA layout: lanes 0-15 hold M=0..15 with
// K = {0..7, 16..23}; lanes 16-31 hold K = {8..15, 24..31}.  `A` is row-major
// f16 with row stride `stride` (f16 elements); `m0` = first row, `kc` = f16
// offset of the 32-wide K chunk.
__device__ __forceinline__ v16h load_a_frag(const _Float16* A, int m0, int kc,
                                            int stride) {
  const int lane = threadIdx.x & 31;
  const int m    = m0 + (lane & 15);
  const int hf   = lane >> 4;
  const _Float16* p = A + (size_t)m * stride + kc + hf * 8;
  v8h lo = *(const v8h*)(p);
  v8h hi = *(const v8h*)(p + 16);
  v16h r;
#pragma unroll
  for (int i = 0; i < 8; ++i) { r[i] = lo[i]; r[i + 8] = hi[i]; }
  return r;
}

// B fragment (32x16 f16, K x N) from an N-major LDS buffer Bt[n][k]
// (row stride `stride` f16).  lanes 0-15: N=n, K={0..7,16..23}; lanes 16-31: K={8..15,24..31}.
__device__ __forceinline__ v16h load_b_frag(const _Float16* Bt, int n0, int kc,
                                            int stride) {
  const int lane = threadIdx.x & 31;
  const int n    = n0 + (lane & 15);
  const int hf   = lane >> 4;
  const _Float16* p = Bt + (size_t)n * stride + kc + hf * 8;
  v8h lo = *(const v8h*)(p);
  v8h hi = *(const v8h*)(p + 16);
  v16h r;
#pragma unroll
  for (int i = 0; i < 8; ++i) { r[i] = lo[i]; r[i + 8] = hi[i]; }
  return r;
}

// ---------------------------------------------------------------------------
// f32 -> f16 conversion
// ---------------------------------------------------------------------------
__global__ void __launch_bounds__(256)
f32_to_f16_kernel(const float* __restrict__ X, _Float16* __restrict__ Y,
                  size_t n) {
  size_t i = (size_t)blockIdx.x * 256 + threadIdx.x;
  const size_t stride = (size_t)gridDim.x * 256;
  for (; i < n; i += stride) Y[i] = (_Float16)X[i];
}

// ---------------------------------------------------------------------------
// Generic WMMA GEMM:  C[M,N] = A[M,K] * B[K,N]   (A,B f16 row-major, packed)
// block tile 256x128x64, 256 threads = 8 waves; wave w owns rows [32w,32w+32)
// (16 v8f accumulators) so each B fragment feeds two WMMAs.
// Double-buffered LDS (96 KB dynamic): async-stage tile k+1 while computing
// tile k; one s_wait_asynccnt 0 + barrier per K-step.
// M % 256 == 0, K % 64 == 0 required; N handled with bounds checks.
// ---------------------------------------------------------------------------
template <bool OUT_F16>
__global__ void __launch_bounds__(256)
gemm_wmma_kernel(const _Float16* __restrict__ A, const _Float16* __restrict__ B,
                 void* __restrict__ Cv, int M, int N, int K) {
  extern __shared__ char gsmem[];
  _Float16* AsBuf = (_Float16*)gsmem;          // 2 x 256*64 (64 KB)
  _Float16* BtBuf = AsBuf + 2 * 256 * 64;      // 2 x 128*64 (32 KB)

  const int tid  = threadIdx.x;
  const int wave = tid >> 5;
  const int lane = tid & 31;
  const int hf   = lane >> 4;
  const int nlo  = lane & 15;
  const int n0   = blockIdx.x * 128;
  const int m0   = blockIdx.y * 256;
  const bool nfull = (n0 + 128) <= N;

  v8f acc[16];
#pragma unroll
  for (int i = 0; i < 16; ++i) acc[i] = vzero8();

  // per-thread B tile-copy coordinates
  const int bk = tid >> 2;            // 0..63      B row (k)
  const int bn = (tid & 3) * 32;      // 0/32/64/96 B col (n)

  // ---- stage A tile 256x64: thread t owns row t (64 f16 = 8x async b128) --
  auto stageA = [&](int buf, int k0) {
    const _Float16* src = A + (size_t)(m0 + tid) * K + k0;
    _Float16* dst = AsBuf + buf * (256 * 64) + tid * 64;
#pragma unroll
    for (int c = 0; c < 64; c += 8) async_copy_b128(dst + c, src + c);
  };
  // ---- stage B tile 64x128, transposed into Bt[n][k] ----------------------
  auto stageB = [&](int buf, int k0) {
    const _Float16* src = B + (size_t)(k0 + bk) * N + n0 + bn;
    _Float16* dst = BtBuf + buf * (128 * 64);
    if (nfull) {
#pragma unroll
      for (int c = 0; c < 32; c += 8) {
        v8h v = *(const v8h*)(src + c);
#pragma unroll
        for (int j = 0; j < 8; ++j) dst[(bn + c + j) * 64 + bk] = v[j];
      }
      if (k0 + 64 < K) __builtin_prefetch(src + (size_t)64 * N, 0, 0);
    } else {
#pragma unroll
      for (int j = 0; j < 32; ++j) {
        _Float16 v = (_Float16)0.0f;
        if (n0 + bn + j < N) v = src[j];
        dst[(bn + j) * 64 + bk] = v;
      }
    }
  };

  // prologue: fill buffer 0
  stageA(0, 0);
  stageB(0, 0);
  wait_async0();
  __syncthreads();

  int buf = 0;
  for (int k0 = 0; k0 < K; k0 += 64) {
    // kick off next tile into the other buffer while we compute
    if (k0 + 64 < K) { stageA(buf ^ 1, k0 + 64); stageB(buf ^ 1, k0 + 64); }

    const _Float16* As = AsBuf + buf * (256 * 64);
    const _Float16* Bt = BtBuf + buf * (128 * 64);
#pragma unroll
    for (int kc = 0; kc < 64; kc += 32) {
      v16h af0 = load_a_frag(As, wave * 32, kc, 64);
      v16h af1 = load_a_frag(As, wave * 32 + 16, kc, 64);
#pragma unroll
      for (int ct = 0; ct < 8; ++ct) {
        v16h bf = load_b_frag(Bt, ct * 16, kc, 64);
        acc[ct]     = wmma_f16(af0, bf, acc[ct]);
        acc[8 + ct] = wmma_f16(af1, bf, acc[8 + ct]);
      }
    }
    wait_async0();     // next-tile async copies landed in LDS
    __syncthreads();   // everyone done reading cur + writing nxt
    buf ^= 1;
  }

  // ---- epilogue: C tile layout: VGPR g -> M = g + hf*8, lanes hold N ----
#pragma unroll
  for (int sub = 0; sub < 2; ++sub) {
#pragma unroll
    for (int ct = 0; ct < 8; ++ct) {
#pragma unroll
      for (int g = 0; g < 8; ++g) {
        const int row = m0 + wave * 32 + sub * 16 + g + hf * 8;
        const int col = n0 + ct * 16 + nlo;
        if (col < N) {
          if (OUT_F16)
            ((_Float16*)Cv)[(size_t)row * N + col] = (_Float16)acc[sub * 8 + ct][g];
          else
            ((float*)Cv)[(size_t)row * N + col] = acc[sub * 8 + ct][g];
        }
      }
    }
  }
}

// ---------------------------------------------------------------------------
// RMSNorm rows: Y[r][0..n) = X[r][0..n) * rsqrt(mean(x^2)+eps) * W, f16 out.
// X has row stride `strideIn` (>= n).
// ---------------------------------------------------------------------------
__global__ void __launch_bounds__(256)
rmsnorm_kernel(const float* __restrict__ X, const float* __restrict__ W,
               _Float16* __restrict__ Y, int n, int strideIn) {
  __shared__ float red[256];
  const int row = blockIdx.x;
  const float* x = X + (size_t)row * strideIn;
  float s = 0.0f;
  for (int j = threadIdx.x; j < n; j += 256) { float v = x[j]; s += v * v; }
  red[threadIdx.x] = s;
  __syncthreads();
  for (int off = 128; off > 0; off >>= 1) {
    if (threadIdx.x < off) red[threadIdx.x] += red[threadIdx.x + off];
    __syncthreads();
  }
  const float scale = rsqrtf(red[0] / (float)n + kEps);
  _Float16* y = Y + (size_t)row * n;
  for (int j = threadIdx.x; j < n; j += 256)
    y[j] = (_Float16)(x[j] * scale * W[j]);
}

// ---------------------------------------------------------------------------
// Q post-process: deinterleave+RoPE on last 64 dims, fold in SCALING, f16 out.
// grid (T, H), block 192.
// ---------------------------------------------------------------------------
__global__ void __launch_bounds__(192)
q_post_kernel(const float* __restrict__ Qraw, const float* __restrict__ C,
              const float* __restrict__ S, _Float16* __restrict__ Qf) {
  const int t = blockIdx.x, h = blockIdx.y, j = threadIdx.x;
  const float* q = Qraw + ((size_t)t * kH + h) * kDQK;
  float v;
  if (j < kDNope) {
    v = q[j];
  } else {
    const int p = j - kDNope;
    float di, rot;
    if (p < 32) { di = q[kDNope + 2 * p];            rot = -q[kDNope + 2 * p + 1]; }
    else        { di = q[kDNope + 2 * (p - 32) + 1]; rot =  q[kDNope + 2 * (p - 32)]; }
    v = di * C[t * kDRope + p] + rot * S[t * kDRope + p];
  }
  Qf[((size_t)t * kH + h) * kDQK + j] = (_Float16)(v * kScaling);
}

// k_pe RoPE: ckv cols [512,576) -> f16 [T][64].  grid T, block 64.
__global__ void __launch_bounds__(64)
kpe_rope_kernel(const float* __restrict__ CKV, const float* __restrict__ C,
                const float* __restrict__ S, _Float16* __restrict__ KPE) {
  const int t = blockIdx.x, p = threadIdx.x;
  const float* k = CKV + (size_t)t * (kKVLora + kDRope) + kKVLora;
  float di, rot;
  if (p < 32) { di = k[2 * p];            rot = -k[2 * p + 1]; }
  else        { di = k[2 * (p - 32) + 1]; rot =  k[2 * (p - 32)]; }
  KPE[t * kDRope + p] = (_Float16)(di * C[t * kDRope + p] + rot * S[t * kDRope + p]);
}

// ---------------------------------------------------------------------------
// Causal flash attention, one (head, 128-row q-tile) per block, 8 waves.
// Q [T][H*192] (pre-scaled), KV [T][H*256] (k_nope | v), KPE [T][64],
// out attn f16 [T][H*128].  Dynamic LDS ~226 KB (WGP has 320 KB).
// ---------------------------------------------------------------------------
__global__ void __launch_bounds__(256)
mla_attention_kernel(const _Float16* __restrict__ Q,
                     const _Float16* __restrict__ KV,
                     const _Float16* __restrict__ KPE,
                     _Float16* __restrict__ Oout) {
  extern __shared__ char smem_raw[];
  _Float16* Qs = (_Float16*)smem_raw;        // 128 x 192
  _Float16* Ks = Qs + 128 * 192;             // 128 x 192  ([n][d] = Bt layout)
  _Float16* Vt = Ks + 128 * 192;             // 128 x 128  ([dv][n])
  _Float16* Ps = Vt + 128 * 128;             // 128 x 128  ([m][n], f16 probs)
  float*    Sf = (float*)(Ps + 128 * 128);   // 128 x 128 f32 scores
  float* rowM  = Sf + 128 * 128;
  float* rowL  = rowM + 128;
  float* rowA  = rowL + 128;

  const int tid  = threadIdx.x;
  const int wave = tid >> 5;
  const int lane = tid & 31;
  const int hf   = lane >> 4;
  const int nlo  = lane & 15;
  const int qt   = blockIdx.x;
  const int head = blockIdx.y;
  const int q0   = qt * 128;

  // stage Q tile asynchronously (192 f16 per row = 24 chunks of 8)
  for (int i = tid; i < 128 * 24; i += 256) {
    const int r = i / 24, c = (i % 24) * 8;
    async_copy_b128(Qs + r * 192 + c,
                    Q + ((size_t)(q0 + r) * kH + head) * kDQK + c);
  }
  v8f o[8];
#pragma unroll
  for (int i = 0; i < 8; ++i) o[i] = vzero8();
  if (tid < 128) { rowM[tid] = -3.0e38f; rowL[tid] = 0.0f; }
  __syncthreads();

  for (int kt = 0; kt <= qt; ++kt) {
    const int k0t = kt * 128;
    // ---- stage K (k_nope | k_pe) async, V^T via VALU transpose ----
    for (int i = tid; i < 2048; i += 256) {          // k_nope: 128x128
      const int r = i >> 4, c = (i & 15) * 8;
      async_copy_b128(Ks + r * 192 + c,
                      KV + ((size_t)(k0t + r) * kH + head) * 256 + c);
    }
    for (int i = tid; i < 1024; i += 256) {          // k_pe: 128x64 (head-shared)
      const int r = i >> 3, c = (i & 7) * 8;
      async_copy_b128(Ks + r * 192 + 128 + c,
                      KPE + (size_t)(k0t + r) * kDRope + c);
    }
    for (int i = tid; i < 2048; i += 256) {          // V transpose -> Vt[dv][n]
      const int n = i >> 4, c = (i & 15) * 8;
      v8h v = *(const v8h*)(KV + ((size_t)(k0t + n) * kH + head) * 256 + 128 + c);
#pragma unroll
      for (int j = 0; j < 8; ++j) Vt[(c + j) * 128 + n] = v[j];
    }
    wait_async0();
    __syncthreads();

    // ---- S = Q K^T (per wave: 16 rows x 128 cols, K-dim 192 = 6 chunks) ----
    v16h qa[6];
#pragma unroll
    for (int dk = 0; dk < 6; ++dk) qa[dk] = load_a_frag(Qs, wave * 16, dk * 32, 192);
#pragma unroll
    for (int ct = 0; ct < 8; ++ct) {
      v8f s = vzero8();
#pragma unroll
      for (int dk = 0; dk < 6; ++dk) {
        v16h bf = load_b_frag(Ks, ct * 16, dk * 32, 192);
        s = wmma_f16(qa[dk], bf, s);
      }
#pragma unroll
      for (int g = 0; g < 8; ++g)
        Sf[(wave * 16 + g + hf * 8) * 128 + ct * 16 + nlo] = s[g];
    }
    __syncthreads();

    // ---- online softmax (one row per thread) ----
    if (tid < 128) {
      const int r = tid;
      const float* srow = Sf + r * 128;
      const int limit = (kt == qt) ? r : 127;   // causal
      float mx = -3.0e38f;
      for (int n = 0; n <= limit; ++n) mx = fmaxf(mx, srow[n]);
      const float mold = rowM[r];
      const float mnew = fmaxf(mold, mx);
      const float alpha = __expf(mold - mnew);
      float sum = 0.0f;
      _Float16* prow = Ps + r * 128;
      for (int n = 0; n < 128; ++n) {
        const float p = (n <= limit) ? __expf(srow[n] - mnew) : 0.0f;
        sum += p;
        prow[n] = (_Float16)p;
      }
      rowL[r] = rowL[r] * alpha + sum;
      rowM[r] = mnew;
      rowA[r] = alpha;
    }
    __syncthreads();

    // ---- rescale O, then O += P * V ----
    v16h pf[4];
#pragma unroll
    for (int nk = 0; nk < 4; ++nk) pf[nk] = load_a_frag(Ps, wave * 16, nk * 32, 128);
#pragma unroll
    for (int ct = 0; ct < 8; ++ct) {
#pragma unroll
      for (int g = 0; g < 8; ++g) o[ct][g] *= rowA[wave * 16 + g + hf * 8];
#pragma unroll
      for (int nk = 0; nk < 4; ++nk) {
        v16h vf = load_b_frag(Vt, ct * 16, nk * 32, 128);
        o[ct] = wmma_f16(pf[nk], vf, o[ct]);
      }
    }
    __syncthreads();
  }

  // ---- normalize and store ----
#pragma unroll
  for (int ct = 0; ct < 8; ++ct) {
#pragma unroll
    for (int g = 0; g < 8; ++g) {
      const int r = wave * 16 + g + hf * 8;
      const float inv = 1.0f / rowL[r];
      Oout[((size_t)(q0 + r) * kH + head) * kDV + ct * 16 + nlo] =
          (_Float16)(o[ct][g] * inv);
    }
  }
}

// ---------------------------------------------------------------------------
// Host launcher
// ---------------------------------------------------------------------------
extern "C" void kernel_launch(void* const* d_in, const int* in_sizes, int n_in,
                              void* d_out, int out_size, void* d_ws, size_t ws_size,
                              hipStream_t stream) {
  (void)in_sizes; (void)n_in; (void)out_size; (void)ws_size;
  const float* hs    = (const float*)d_in[0];
  const float* cosT  = (const float*)d_in[1];
  const float* sinT  = (const float*)d_in[2];
  const float* wq_a  = (const float*)d_in[3];
  const float* q_ln  = (const float*)d_in[4];
  const float* wq_b  = (const float*)d_in[5];
  const float* wkv_a = (const float*)d_in[6];
  const float* kv_ln = (const float*)d_in[7];
  const float* wkv_b = (const float*)d_in[8];
  const float* wo    = (const float*)d_in[9];
  float* out = (float*)d_out;

  // ---- workspace carve ----
  char* p = (char*)d_ws;
  auto alloc16 = [&](size_t elems) {
    _Float16* r = (_Float16*)p; p += ((elems * 2 + 255) / 256) * 256; return r;
  };
  auto alloc32 = [&](size_t elems) {
    float* r = (float*)p; p += ((elems * 4 + 255) / 256) * 256; return r;
  };
  _Float16* hs16   = alloc16((size_t)kT * kHidden);
  _Float16* wqa16  = alloc16((size_t)kHidden * kQLora);
  _Float16* wqb16  = alloc16((size_t)kQLora * kH * kDQK);
  _Float16* wkva16 = alloc16((size_t)kHidden * (kKVLora + kDRope));
  _Float16* wkvb16 = alloc16((size_t)kKVLora * kH * (kDNope + kDV));
  _Float16* wo16   = alloc16((size_t)kH * kDV * kHidden);
  float*    qa32   = alloc32((size_t)kT * kQLora);
  _Float16* qan16  = alloc16((size_t)kT * kQLora);
  float*    ckv32  = alloc32((size_t)kT * (kKVLora + kDRope));
  _Float16* cn16   = alloc16((size_t)kT * kKVLora);
  _Float16* kpe16  = alloc16((size_t)kT * kDRope);
  float*    qraw32 = alloc32((size_t)kT * kH * kDQK);
  _Float16* qf16   = alloc16((size_t)kT * kH * kDQK);
  _Float16* kv16   = alloc16((size_t)kT * kH * (kDNope + kDV));
  _Float16* attn16 = alloc16((size_t)kT * kH * kDV);

  auto cvt = [&](const float* src, _Float16* dst, size_t n) {
    int grid = (int)((n + 256 * 8 - 1) / (256 * 8));
    if (grid > 8192) grid = 8192;
    f32_to_f16_kernel<<<grid, 256, 0, stream>>>(src, dst, n);
  };
  // 1) f32 -> f16 conversions (activations + weights)
  cvt(hs,    hs16,   (size_t)kT * kHidden);
  cvt(wq_a,  wqa16,  (size_t)kHidden * kQLora);
  cvt(wq_b,  wqb16,  (size_t)kQLora * kH * kDQK);
  cvt(wkv_a, wkva16, (size_t)kHidden * (kKVLora + kDRope));
  cvt(wkv_b, wkvb16, (size_t)kKVLora * kH * (kDNope + kDV));
  cvt(wo,    wo16,   (size_t)kH * kDV * kHidden);

  auto gdim = [](int M, int N) { return dim3((N + 127) / 128, M / 256); };
  const size_t smemGemm = (size_t)(2 * 256 * 64 + 2 * 128 * 64) * 2;  // 96 KB

  // 2) qa = hs @ wq_a       [2048 x 1536], K=5120
  gemm_wmma_kernel<false><<<gdim(kT, kQLora), 256, smemGemm, stream>>>(
      hs16, wqa16, qa32, kT, kQLora, kHidden);
  // 3) ckv = hs @ wkv_a     [2048 x 576], K=5120
  gemm_wmma_kernel<false><<<gdim(kT, kKVLora + kDRope), 256, smemGemm, stream>>>(
      hs16, wkva16, ckv32, kT, kKVLora + kDRope, kHidden);
  // 4) rmsnorm(qa) -> f16
  rmsnorm_kernel<<<kT, 256, 0, stream>>>(qa32, q_ln, qan16, kQLora, kQLora);
  // 5) rmsnorm(ckv[:, :512]) -> f16
  rmsnorm_kernel<<<kT, 256, 0, stream>>>(ckv32, kv_ln, cn16, kKVLora,
                                         kKVLora + kDRope);
  // 6) RoPE on k_pe
  kpe_rope_kernel<<<kT, 64, 0, stream>>>(ckv32, cosT, sinT, kpe16);
  // 7) q = qa_n @ wq_b      [2048 x 6144], K=1536
  gemm_wmma_kernel<false><<<gdim(kT, kH * kDQK), 256, smemGemm, stream>>>(
      qan16, wqb16, qraw32, kT, kH * kDQK, kQLora);
  // 8) RoPE + scale on q -> f16
  q_post_kernel<<<dim3(kT, kH), 192, 0, stream>>>(qraw32, cosT, sinT, qf16);
  // 9) kv = c_n @ wkv_b     [2048 x 8192], K=512 -> f16 directly
  gemm_wmma_kernel<true><<<gdim(kT, kH * (kDNope + kDV)), 256, smemGemm, stream>>>(
      cn16, wkvb16, kv16, kT, kH * (kDNope + kDV), kKVLora);
  // 10) attention
  {
    const size_t smem = (size_t)(128 * 192 * 2) * 2   // Qs + Ks
                      + (size_t)(128 * 128 * 2) * 2   // Vt + Ps
                      + (size_t)(128 * 128 * 4)       // Sf
                      + (size_t)(3 * 128 * 4);        // rowM/L/A
    mla_attention_kernel<<<dim3(kT / 128, kH), 256, smem, stream>>>(
        qf16, kv16, kpe16, attn16);
  }
  // 11) out = attn @ wo     [2048 x 5120], K=4096 -> f32 d_out
  gemm_wmma_kernel<false><<<gdim(kT, kHidden), 256, smemGemm, stream>>>(
      attn16, wo16, out, kT, kHidden, kH * kDV);
}
